// FC_GPFA_53077205844444
// MI455X (gfx1250) — matrix operand order repaired and checked by reading, not compiled, not run on hardware
//
#include <hip/hip_runtime.h>
#include <hip/hip_bf16.h>
#include <stdint.h>

// ---- problem constants (from reference setup) ----
#define NTRIAL  8
#define NNEUR   100
#define NTIME   256
#define NT_FULL 300
#define NTAU    50
#define NLAT    8
#define NPADD   44
#define LRATE   0.2f
#define JITTERV 1e-3f
#define ZT      1e-5f
#define NITER   5

typedef __attribute__((ext_vector_type(16))) __bf16          v16bf;
typedef __attribute__((ext_vector_type(8)))  float           v8f;
typedef __attribute__((ext_vector_type(2)))  float           v2f;
typedef __attribute__((ext_vector_type(16))) unsigned short  v16u;
typedef __attribute__((ext_vector_type(8)))  unsigned short  u16x8;
typedef __attribute__((ext_vector_type(4)))  unsigned int    u32x4;
typedef __attribute__((ext_vector_type(8)))  int             i32x8;
typedef __attribute__((ext_vector_type(4)))  int             i32x4;

__device__ __forceinline__ unsigned short f2bf(float x) {
  unsigned int u = __float_as_uint(x);
  u += 0x7fffu + ((u >> 16) & 1u);          // round-to-nearest-even
  return (unsigned short)(u >> 16);
}

// D = A(bf16 16x32) * B(bf16 32x16) + C(f32 16x16), wave32 WMMA
__device__ __forceinline__ v8f wmma_bf16(const v16u a, const v16u b, v8f c) {
  return __builtin_amdgcn_wmma_f32_16x16x32_bf16(
      false, __builtin_bit_cast(v16bf, a),
      false, __builtin_bit_cast(v16bf, b),
      (short)0, c, false, false);
}

__device__ __forceinline__ v16u cat8(u16x8 lo, u16x8 hi) {
  return __builtin_shufflevector(lo, hi, 0, 1, 2, 3, 4, 5, 6, 7, 8, 9, 10, 11, 12, 13, 14, 15);
}

#if __has_builtin(__builtin_amdgcn_wmma_f32_16x16x4_f32)
#define HAVE_F32_WMMA 1
// full-precision f32 WMMA (16x16x4), chained x4 -> one 16x16x16 tile GEMM
__device__ __forceinline__ v8f wmma_f32x4(v2f a, v2f b, v8f c) {
  return __builtin_amdgcn_wmma_f32_16x16x4_f32(false, a, false, b, (short)0, c, false, false);
}
// C += A(16x16, row stride lda) * B(16x16, row stride ldb), operands in LDS
__device__ __forceinline__ v8f tile_mm16(const float* __restrict__ A, int lda,
                                         const float* __restrict__ B, int ldb, int lane) {
  const int ml = lane & 15;
  const int g2 = (lane >> 4) << 1;              // 0 or 2 (K sub-offset per lane group)
  v8f c = {0.f, 0.f, 0.f, 0.f, 0.f, 0.f, 0.f, 0.f};
#pragma unroll
  for (int kk = 0; kk < 16; kk += 4) {
    v2f a, b;
    a[0] = A[ml * lda + kk + g2];
    a[1] = A[ml * lda + kk + g2 + 1];
    b[0] = B[(kk + g2) * ldb + ml];
    b[1] = B[(kk + g2 + 1) * ldb + ml];
    c = wmma_f32x4(a, b, c);
  }
  return c;
}
#endif

// ------------------------------------------------------------------
// init: mu = 0 (in d_out), zero bf16 G-slab (incl. i-padding 100..127)
// ------------------------------------------------------------------
__global__ void k_init(float* mu, unsigned int* slab_u32, long n_slab_u32) {
  long i = (long)blockIdx.x * blockDim.x + threadIdx.x;
  long stride = (long)gridDim.x * blockDim.x;
  for (long k = i; k < NTRIAL * NLAT * NTIME; k += stride) mu[k] = 0.f;
  for (long k = i; k < n_slab_u32; k += stride) slab_u32[k] = 0u;
}

// ------------------------------------------------------------------
// In-LDS inversion of P = src + diag (256x256 fp32): blocked in-place
// Gauss-Jordan. 320KB LDS/WGP keeps the whole 257-padded matrix
// resident. 16x16 pivot blocks inverted in-registers by wave0 (shfl,
// no barriers); row/col panels and rank-16 trailing updates run as
// V_WMMA_F32_16X16X4_F32 tile GEMMs (4 barriers per block step).
// GJ pivots give logabsdet for free.
//
// mode==0: out = sign*P^-1 stored to global (used once, for inv_K).
// mode==1: the inverse NEVER leaves LDS. Fused epilogue computes
//   dh_out[b][t]  = diag(sign*P^-1)            (for lambda)
//   trh_out[b]    = tr(invK @ sign*P^-1)       (for the loss)
//   mu[b]        -= LR * sign * (P^-1 @ grad)  (the solve/update)
// ld_out[b] = -sum log|pivot| = logabsdet(sign*P^-1)
// ------------------------------------------------------------------
__global__ __launch_bounds__(1024) void k_gj(
    const float* __restrict__ src, const float* __restrict__ diagv,
    float diag_const, float* __restrict__ out, float* __restrict__ ld_out,
    float sign, int mode, float* __restrict__ dh_out, float* __restrict__ trh_out,
    const float* __restrict__ grad_in, float* __restrict__ mu_io) {
  __shared__ float M[256 * 257];            // 257 pad: stride % 64 banks == 1
  __shared__ float Dm[16 * 17];             // inverted diagonal block
  __shared__ float gbuf[256];
  __shared__ float sc[2];
  const int tid = threadIdx.x;
  const int b   = blockIdx.x;

  for (int f = tid; f < 65536; f += 1024) {
    int i = f >> 8, j = f & 255;
    float v = src[f];
    if (i == j) v += mode ? diagv[b * 256 + i] : diag_const;
    M[i * 257 + j] = v;
  }
  if (tid == 0) sc[1] = 0.f;

#if defined(HAVE_F32_WMMA)
  const int lane = tid & 31;
  const int wid  = __builtin_amdgcn_readfirstlane(tid >> 5);  // keep scheduling in SALU
  const int ml = lane & 15, g = lane >> 4;
  for (int kb = 0; kb < 16; ++kb) {
    __syncthreads();
    // --- 16x16 pivot-block inverse, in registers of wave0 (no barriers) ---
    if (wid == 0) {
      float r[16];
      float la = 0.f;
#pragma unroll
      for (int j = 0; j < 16; ++j)
        r[j] = (lane < 16) ? M[(kb * 16 + lane) * 257 + kb * 16 + j] : 0.f;
#pragma unroll
      for (int k = 0; k < 16; ++k) {
        float prow[16];
#pragma unroll
        for (int j = 0; j < 16; ++j) prow[j] = __shfl(r[j], k);
        float p = prow[k], ip = 1.f / p;
        la += logf(fabsf(p));
        if (lane == k) {
#pragma unroll
          for (int j = 0; j < 16; ++j) r[j] = (j == k) ? ip : prow[j] * ip;
        } else {
          float fm = r[k] * ip;
#pragma unroll
          for (int j = 0; j < 16; ++j) r[j] = (j == k) ? (-fm) : (r[j] - fm * prow[j]);
        }
      }
      if (lane < 16) {
#pragma unroll
        for (int j = 0; j < 16; ++j) Dm[lane * 17 + j] = r[j];
        if (lane == 0) sc[1] += la;
      }
    }
    __syncthreads();
    // --- row panel: M[kb][bj] := Dinv @ M[kb][bj] ---
    for (int tt = wid; tt < 15; tt += 32) {
      int bj = tt + (tt >= kb);
      float* T = &M[(kb * 16) * 257 + bj * 16];
      v8f c = tile_mm16(&Dm[0], 17, T, 257, lane);
#pragma unroll
      for (int r8 = 0; r8 < 8; ++r8) T[(r8 + 8 * g) * 257 + ml] = c[r8];
    }
    __syncthreads();
    // --- trailing rank-16 update: M[bi][bj] -= M[bi][kb] @ M[kb][bj] ---
    for (int tt = wid; tt < 225; tt += 32) {
      int ti = tt / 15, tj = tt % 15;
      int bi = ti + (ti >= kb), bj = tj + (tj >= kb);
      const float* Ac = &M[(bi * 16) * 257 + kb * 16];
      const float* Br = &M[(kb * 16) * 257 + bj * 16];
      float* Cp = &M[(bi * 16) * 257 + bj * 16];
      v8f c = tile_mm16(Ac, 257, Br, 257, lane);
#pragma unroll
      for (int r8 = 0; r8 < 8; ++r8) {
        int off = (r8 + 8 * g) * 257 + ml;
        Cp[off] = Cp[off] - c[r8];
      }
    }
    __syncthreads();
    // --- column panel: M[bi][kb] := -M[bi][kb] @ Dinv ; diag := Dinv ---
    for (int tt = wid; tt < 15; tt += 32) {
      int bi = tt + (tt >= kb);
      float* T = &M[(bi * 16) * 257 + kb * 16];
      v8f c = tile_mm16(T, 257, &Dm[0], 17, lane);
#pragma unroll
      for (int r8 = 0; r8 < 8; ++r8) T[(r8 + 8 * g) * 257 + ml] = -c[r8];
    }
    if (wid == 31 && lane < 16) {
#pragma unroll
      for (int j = 0; j < 16; ++j)
        M[(kb * 16 + lane) * 257 + kb * 16 + j] = Dm[lane * 17 + j];
    }
  }
  __syncthreads();
#else
  // scalar fallback (no f32 WMMA builtin available)
  __shared__ float colk[256];
  for (int k = 0; k < 256; ++k) {
    __syncthreads();
    if (tid < 256) colk[tid] = M[tid * 257 + k];
    if (tid == 0) {
      float p = M[k * 257 + k];
      sc[0] = 1.f / p;
      sc[1] += logf(fabsf(p));
    }
    __syncthreads();
    const float ip = sc[0];
    for (int f = tid; f < 65536; f += 1024) {
      int i = f >> 8, j = f & 255;
      if (i != k) {
        float v = (j == k) ? (-colk[i] * ip)
                           : (M[i * 257 + j] - colk[i] * (ip * M[k * 257 + j]));
        M[i * 257 + j] = v;
      }
    }
    __syncthreads();
    if (tid < 256) M[k * 257 + tid] = (tid == k) ? ip : M[k * 257 + tid] * ip;
  }
  __syncthreads();
#endif

  if (out) {   // mode 0: materialize the inverse (inv_K only)
    for (int f = tid; f < 65536; f += 1024)
      out[(size_t)b * 65536 + f] = sign * M[(f >> 8) * 257 + (f & 255)];
  }
  if (mode) {  // fused epilogue: inverse stays in LDS
    if (tid < 256) {
      dh_out[b * 256 + tid] = sign * M[tid * 257 + tid];
      gbuf[tid] = grad_in[(size_t)b * 256 + tid];
    }
    if (tid == 0) sc[0] = 0.f;
    __syncthreads();
    float tr = 0.f;                          // tr(invK @ hinv), src == invK here
    for (int f = tid; f < 65536; f += 1024) {
      int t = f >> 8, u = f & 255;
      tr += src[(size_t)t * 256 + u] * M[u * 257 + t];
    }
    atomicAdd(&sc[0], tr);
    __syncthreads();
    if (tid == 0) trh_out[b] = sign * sc[0];
    if (tid < 256) {                         // mu -= LR * hinv_new @ grad
      const float* mr = &M[tid * 257];
      float ud = 0.f;
      for (int u = 0; u < 256; ++u) ud += mr[u] * gbuf[u];
      mu_io[(size_t)b * 256 + tid] -= LRATE * (sign * ud);
    }
  }
  if (tid == 0) ld_out[b] = -sc[1];
}

// ------------------------------------------------------------------
// Iteration-0 stats for hinv_0 = K + jitter*I (never materialized):
// dh = diag(K)+jitter ; trh[b] = tr(invK @ (K+jI)) ; ld_hinv = -ld_invK
// ------------------------------------------------------------------
__global__ void k_stats0(const float* __restrict__ Kmat, const float* __restrict__ invK,
                         float* __restrict__ ld_hinv, const float* __restrict__ ld_invK,
                         float* __restrict__ dh, float* __restrict__ trh) {
  __shared__ float red[256];
  const int b = blockIdx.x, t = threadIdx.x;
  dh[b * 256 + t] = Kmat[(size_t)t * 256 + t] + JITTERV;
  float tr = 0.f;
  for (int u = 0; u < 256; ++u) {
    float kv = Kmat[(size_t)u * 256 + t] + ((u == t) ? JITTERV : 0.f);
    tr += invK[(size_t)t * 256 + u] * kv;
  }
  red[t] = tr;
  __syncthreads();
  for (int st = 128; st > 0; st >>= 1) { if (t < st) red[t] += red[t + st]; __syncthreads(); }
  if (t == 0) {
    trh[b] = red[0];
    ld_hinv[b] = -ld_invK[0];
  }
}

// ------------------------------------------------------------------
// Conv as GEMM: G[t(256) x j(100)] = A[t x tau(64)] * Keff[tau x j]
// per (m,i) workgroup. Kernel slab staged into LDS via TDM.
// kerT is stored column(j)-major so each B fragment is two 16-byte
// LDS reads (2x ds_load_b128) instead of 16 scalar loads.
// Output: thresholded bf16 slab laid out [j][t][i-pad128].
// ------------------------------------------------------------------
__global__ __launch_bounds__(256) void k_conv(
    const float* __restrict__ raw, const float* __restrict__ ker,
    unsigned short* __restrict__ gslab, int m) {
  __shared__ float kslab[NTAU * NNEUR];                      // 20000 B, TDM target
  __shared__ __align__(32) unsigned short kerT[112 * 64];    // Keff^T [j][tau]
  __shared__ unsigned short rawbf[304];
  const int i   = blockIdx.x;
  const int tid = threadIdx.x;

#if defined(__gfx1250__) && __has_builtin(__builtin_amdgcn_tensor_load_to_lds) && \
    __has_builtin(__builtin_amdgcn_s_wait_tensorcnt)
  if (tid < 32) {  // one wave issues the TDM descriptor (EXEC ignored by TDM)
    unsigned long long ga = (unsigned long long)(uintptr_t)(ker + (size_t)i * NTAU * NNEUR);
    unsigned int lds_off  = (unsigned int)(uintptr_t)(&kslab[0]);
    u32x4 g0;
    g0[0] = 1u;                         // count=1 (valid descriptor)
    g0[1] = lds_off;                    // lds_addr
    g0[2] = (unsigned int)ga;           // global_addr[31:0]
    g0[3] = (unsigned int)(((ga >> 32) & 0x01FFFFFFull) | 0x80000000u); // addr[56:32] | type=2
    i32x8 g1;
    g1[0] = (int)(2u << 16);              // data_size = 4B
    g1[1] = (int)((unsigned)NNEUR << 16); // tensor_dim0 = 100
    g1[2] = (int)((unsigned)NTAU << 16);  // tensor_dim1 = 50
    g1[3] = (int)((unsigned)NNEUR << 16); // tile_dim0 = 100
    g1[4] = (int)(unsigned)NTAU;          // tile_dim1 = 50
    g1[5] = (int)NNEUR;                   // tensor_dim0_stride = 100
    g1[6] = 0; g1[7] = 0;
    i32x4 z4; z4[0] = z4[1] = z4[2] = z4[3] = 0;
#if __clang_major__ >= 23
    i32x8 z8; for (int q = 0; q < 8; ++q) z8[q] = 0;
    __builtin_amdgcn_tensor_load_to_lds(g0, g1, z4, z4, z8, 0);
#else
    __builtin_amdgcn_tensor_load_to_lds(g0, g1, z4, z4, 0);
#endif
    __builtin_amdgcn_s_wait_tensorcnt(0);
  }
  __syncthreads();
#else
  for (int idx = tid; idx < NTAU * NNEUR; idx += 256)
    kslab[idx] = ker[(size_t)i * NTAU * NNEUR + idx];
  __syncthreads();
#endif

  // Keff^T[j][0]=0 (reference zero-pads kernel), Keff^T[j][k]=ker[i][k-1][j]
  for (int idx = tid; idx < 112 * 64; idx += 256) {
    int j = idx >> 6, tau = idx & 63;
    float v = (tau >= 1 && tau <= NTAU && j < NNEUR) ? kslab[(tau - 1) * NNEUR + j] : 0.f;
    kerT[idx] = f2bf(v);
  }
  for (int idx = tid; idx < 304; idx += 256) {
    float v = (idx < NT_FULL) ? raw[((size_t)m * NNEUR + i) * NT_FULL + idx] : 0.f;
    rawbf[idx] = f2bf(v);
  }
  __syncthreads();

  const int wave = tid >> 5, lane = tid & 31, lm = lane & 15;
  const int hi8 = (lane >= 16) ? 8 : 0, hi16 = hi8 * 2;

  for (int mt = wave; mt < 16; mt += 8) {          // 16 t-tiles over 8 waves
    const int tloc = mt * 16 + lm;
    v16u a0, a1;                                    // Toeplitz gather (scalar, reused 7x)
#pragma unroll
    for (int e = 0; e < 16; ++e) {                  // 16-bit A layout: K = (e<8?e:e+8)+hi8
      int kk   = ((e < 8) ? e : e + 8) + hi8;
      int idx0 = NPADD + tloc - kk;                 // tau chunk 0: 0..31
      int tau1 = kk + 32;                           // tau chunk 1: 32..63
      int idx1 = NPADD + tloc - tau1;
      a0[e] = (idx0 >= 0) ? rawbf[idx0] : (unsigned short)0;
      a1[e] = (tau1 <= NTAU && idx1 >= 0) ? rawbf[idx1] : (unsigned short)0;
    }
    for (int jt = 0; jt < 7; ++jt) {               // 112-wide j (100 valid)
      const int j = jt * 16 + lm;
      // B fragment: runs [hi8 .. +7] and [hi8+16 .. +23] per chunk (16B aligned)
      v16u b0 = cat8(*(const u16x8*)(kerT + (size_t)j * 64 + hi8), *(const u16x8*)(kerT + (size_t)j * 64 + hi8 + 16));
      v16u b1 = cat8(*(const u16x8*)(kerT + (size_t)j * 64 + hi8 + 32), *(const u16x8*)(kerT + (size_t)j * 64 + hi8 + 48));
      v8f c = {0.f, 0.f, 0.f, 0.f, 0.f, 0.f, 0.f, 0.f};
      c = wmma_bf16(a0, b0, c);
      c = wmma_bf16(a1, b1, c);
      if (j < NNEUR) {
#pragma unroll
        for (int r = 0; r < 8; ++r) {               // C/D layout: M = r + hi8
          int t = mt * 16 + r + hi8;
          float gv = c[r];
          if (fabsf(gv) < ZT) gv = 0.f;             // reference threshold
          gslab[((size_t)j * NTIME + t) * 128 + i] = f2bf(gv);
        }
      }
    }
  }
}

// ------------------------------------------------------------------
// weights[m,j,l,t] = sum_i G[m,i,j,t] * readout[i,l]   (GEMM K=i pad128)
// A fragments: two 16B global_load_b128 per chunk; B from transposed
// LDS readout (two 16B reads per chunk).
// ------------------------------------------------------------------
__global__ __launch_bounds__(128) void k_weights(
    const unsigned short* __restrict__ gslab, const float* __restrict__ readout,
    float* __restrict__ W, int m) {
  __shared__ __align__(32) unsigned short robfT[16 * 128];   // [l][i]
  const int j = blockIdx.x, tid = threadIdx.x;
  for (int idx = tid; idx < 16 * 128; idx += 128) {
    int l = idx >> 7, i = idx & 127;
    float v = (i < NNEUR && l < NLAT) ? readout[i * NLAT + l] : 0.f;
    robfT[idx] = f2bf(v);
  }
  __syncthreads();
  const int wave = tid >> 5, lane = tid & 31, lm = lane & 15;
  const int hi8 = (lane >= 16) ? 8 : 0, hi16 = hi8 * 2;
  for (int q = 0; q < 4; ++q) {
    const int mt = wave * 4 + q;
    const int tloc = mt * 16 + lm;
    const unsigned short* gp = gslab + ((size_t)j * NTIME + tloc) * 128;
    v8f c = {0.f, 0.f, 0.f, 0.f, 0.f, 0.f, 0.f, 0.f};
    for (int ch = 0; ch < 4; ++ch) {
      // A: element runs [hi8+32ch .. +7] and [hi8+32ch+16 .. +23], both 16B aligned
      u16x8 alo = *(const u16x8*)(gp + hi8 + 32 * ch);
      u16x8 ahi = *(const u16x8*)(gp + hi8 + 32 * ch + 16);
      v16u  a   = cat8(alo, ahi);
      v16u  b   = cat8(*(const u16x8*)(robfT + (size_t)lm * 128 + hi8 + 32 * ch), *(const u16x8*)(robfT + (size_t)lm * 128 + hi8 + 32 * ch + 16));
      c = wmma_bf16(a, b, c);
    }
    if (lm < NLAT) {
#pragma unroll
      for (int r = 0; r < 8; ++r) {
        int t = mt * 16 + r + hi8;
        W[(((size_t)m * NNEUR + j) * NTIME + t) * NLAT + lm] = c[r];
      }
    }
  }
}

// ------------------------------------------------------------------
// lambda = exp(W.mu + bias + 0.5*diag(hinv).W^2); accumulate loss pieces
// ------------------------------------------------------------------
__global__ __launch_bounds__(256) void k_lambda(
    const float* __restrict__ Y, const float* __restrict__ W,
    const float* __restrict__ mu, const float* __restrict__ dh,
    const float* __restrict__ bias, float* __restrict__ lambd,
    float* __restrict__ acc) {
  __shared__ float red[256];
  const int mn = blockIdx.x;
  const int m = mn / NNEUR;
  const int t = threadIdx.x;
  float ll = bias[0], s = 0.f;
  const float* wp = &W[((size_t)mn * NTIME + t) * NLAT];
#pragma unroll
  for (int l = 0; l < NLAT; ++l) {
    float w   = wp[l];
    float muv = mu[((size_t)m * NLAT + l) * NTIME + t];
    float dhv = dh[((size_t)m * NLAT + l) * NTIME + t];
    ll += w * muv;
    s  += dhv * w * w;
  }
  float lam = expf(ll + 0.5f * s);
  lambd[(size_t)mn * NTIME + t] = lam;
  red[t] = Y[(size_t)mn * NTIME + t] * ll;
  __syncthreads();
  for (int st = 128; st > 0; st >>= 1) { if (t < st) red[t] += red[t + st]; __syncthreads(); }
  if (t == 0) atomicAdd(&acc[0], red[0]);
  __syncthreads();
  red[t] = lam;
  __syncthreads();
  for (int st = 128; st > 0; st >>= 1) { if (t < st) red[t] += red[t + st]; __syncthreads(); }
  if (t == 0) atomicAdd(&acc[1], red[0]);
}

// ------------------------------------------------------------------
// grad, w2l, muK.mu; trace + slogdet terms are precomputed per (m,l)
// ------------------------------------------------------------------
__global__ __launch_bounds__(256) void k_grad(
    const float* __restrict__ Y, const float* __restrict__ W,
    const float* __restrict__ mu, const float* __restrict__ lambd,
    const float* __restrict__ invK,
    float* __restrict__ grad, float* __restrict__ w2l, float* __restrict__ acc,
    const float* __restrict__ ld_invK, const float* __restrict__ ld_hinv,
    const float* __restrict__ trh) {
  __shared__ float murow[256];
  __shared__ float red[256];
  const int b = blockIdx.x;          // b = m*NLAT + l
  const int m = b >> 3, l = b & 7;
  const int t = threadIdx.x;
  murow[t] = mu[(size_t)b * NTIME + t];
  __syncthreads();
  float g = 0.f, wl = 0.f;
  for (int n = 0; n < NNEUR; ++n) {
    size_t base = (size_t)(m * NNEUR + n) * NTIME + t;
    float w  = W[base * NLAT + l];
    float yv = Y[base];
    float la = lambd[base];
    g  += w * (yv - la);
    wl += w * w * la;
    if (n + 1 < NNEUR) __builtin_prefetch(&W[(base + NTIME) * NLAT + l], 0, 1);
  }
  float mk = 0.f;
  for (int u = 0; u < 256; ++u) mk += murow[u] * invK[(size_t)t * 256 + u]; // invK symmetric
  grad[(size_t)b * NTIME + t] = g - mk;
  w2l[(size_t)b * NTIME + t]  = wl;

  red[t] = mk * murow[t];
  __syncthreads();
  for (int st = 128; st > 0; st >>= 1) { if (t < st) red[t] += red[t + st]; __syncthreads(); }
  if (t == 0) {
    atomicAdd(&acc[2], red[0]);
    atomicAdd(&acc[3], trh[b]);                    // tr(invK @ hinv[b]), precomputed
    atomicAdd(&acc[4], ld_invK[0] + ld_hinv[b]);   // logabsdet(invK @ hinv[b])
  }
}

__global__ void k_zero_acc(float* acc) { if (threadIdx.x < 8) acc[threadIdx.x] = 0.f; }

__global__ void k_loss(const float* acc, float* loss) {
  if (threadIdx.x == 0 && blockIdx.x == 0)
    loss[0] = acc[0] - acc[1] - 0.5f * acc[2] - 0.5f * acc[3] + 0.5f * acc[4] - (float)NTIME;
}

// ------------------------------------------------------------------
extern "C" void kernel_launch(void* const* d_in, const int* in_sizes, int n_in,
                              void* d_out, int out_size, void* d_ws, size_t ws_size,
                              hipStream_t stream) {
  const float* Y    = (const float*)d_in[0];
  const float* raw  = (const float*)d_in[1];
  const float* ker  = (const float*)d_in[2];
  const float* rdo  = (const float*)d_in[3];
  const float* Kmat = (const float*)d_in[4];
  const float* bias = (const float*)d_in[5];
  (void)in_sizes; (void)n_in; (void)out_size; (void)ws_size; // max_iter fixed at 5

  char* ws = (char*)d_ws;
  float*          invK    = (float*)(ws + 0);         // 262144 B
  float*          ld_invK = (float*)(ws + 262144);    // 1 f
  float*          ld_hinv = (float*)(ws + 262400);    // 64 f
  float*          grad    = (float*)(ws + 262656);    // 16384 f
  float*          w2l     = (float*)(ws + 328192);    // 16384 f
  float*          acc     = (float*)(ws + 393728);    // 8 f
  float*          dh      = (float*)(ws + 393984);    // 16384 f
  float*          trh     = (float*)(ws + 459520);    // 64 f
  unsigned short* gslab   = (unsigned short*)(ws + 459776);   // 100*256*128 bf16
  float*          Wbuf    = (float*)(ws + 7013376);   // 8*100*256*8 f (ends 13566976)

  float* mu    = (float*)d_out;                       // (8,8,256)
  float* lambd = (float*)d_out + 16384;               // (8,100,256)
  float* loss  = (float*)d_out + 16384 + 204800;      // scalar

  k_init<<<2048, 256, 0, stream>>>(mu, (unsigned int*)gslab, (long)(NNEUR * NTIME * 128 / 2));
  k_gj<<<1, 1024, 0, stream>>>(Kmat, nullptr, JITTERV, invK, ld_invK, 1.0f, 0,
                               nullptr, nullptr, nullptr, nullptr);
  k_stats0<<<64, 256, 0, stream>>>(Kmat, invK, ld_hinv, ld_invK, dh, trh);

  for (int m = 0; m < NTRIAL; ++m) {
    k_conv<<<NNEUR, 256, 0, stream>>>(raw, ker, gslab, m);
    k_weights<<<NNEUR, 128, 0, stream>>>(gslab, rdo, Wbuf, m);
  }

  for (int it = 0; it < NITER; ++it) {
    k_zero_acc<<<1, 64, 0, stream>>>(acc);
    k_lambda<<<NTRIAL * NNEUR, 256, 0, stream>>>(Y, Wbuf, mu, dh, bias, lambd, acc);
    k_grad<<<64, 256, 0, stream>>>(Y, Wbuf, mu, lambd, invK, grad, w2l, acc,
                                   ld_invK, ld_hinv, trh);
    k_gj<<<64, 1024, 0, stream>>>(invK, w2l, 0.f, nullptr, ld_hinv, -1.0f, 1,
                                  dh, trh, grad, mu);
    k_loss<<<1, 1, 0, stream>>>(acc, loss);
  }
}
